// SelectiveSSM_23828478558711
// MI455X (gfx1250) — matrix operand order repaired and checked by reading, not compiled, not run on hardware
//
#include <hip/hip_runtime.h>
#include <hip/hip_bf16.h>
#include <math.h>

// ---------------------------------------------------------------------------
// SelectiveSSM (Mamba block) for MI455X / gfx1250, wave32.
//   Big GEMMs (in-proj 4096x4096x1024, out-proj 4096x1024x2048):
//     v_wmma_f32_16x16x32_bf16, 64x64 block tile double-buffered through LDS
//     via GLOBAL_LOAD_ASYNC_TO_LDS_B128 (ASYNCcnt) when the builtin exists,
//     else a plain b128 load -> ds_store fallback.
//   Conv / x_proj / recurrent scan stay f32 VALU (latency-bound, tiny FLOPs).
// ---------------------------------------------------------------------------

#define D_MODEL  1024
#define D_STATE  16
#define D_INNER  2048
#define BATCH    2
#define SEQ      2048
#define ROWS     (BATCH * SEQ)          // 4096
#define N_XZ     (2 * D_INNER)          // 4096
#define N_PROJ   (1 + 2 * D_STATE)      // 33

typedef __attribute__((ext_vector_type(16))) __bf16 v16bf;
typedef __attribute__((ext_vector_type(8)))  float  v8f;

union FragU {
    v16bf v;
    uint4 q[2];
};

// ---- CDNA5 async global->LDS path (probe-guarded) -------------------------
#if defined(__AMDGCN__) && defined(__has_builtin)
# if __has_builtin(__builtin_amdgcn_global_load_async_to_lds_b128)
#  define USE_ASYNC_LDS 1
# endif
#endif
#ifndef USE_ASYNC_LDS
# define USE_ASYNC_LDS 0
#endif

#if USE_ASYNC_LDS
# define AS1 __attribute__((address_space(1)))
# define AS3 __attribute__((address_space(3)))
// builtin signature is vV4i*1 V4i*3 Ii Ii  (GCC-style vector_size(16) int)
typedef int v4i_vs __attribute__((__vector_size__(16)));
#endif

__device__ __forceinline__ void async_wait_le2() {
#if USE_ASYNC_LDS
# if __has_builtin(__builtin_amdgcn_s_wait_asynccnt)
    __builtin_amdgcn_s_wait_asynccnt(2);
# else
    asm volatile("s_wait_asynccnt 0x2" ::: "memory");
# endif
#endif
}

__device__ __forceinline__ void async_wait_0() {
#if USE_ASYNC_LDS
# if __has_builtin(__builtin_amdgcn_s_wait_asynccnt)
    __builtin_amdgcn_s_wait_asynccnt(0);
# else
    asm volatile("s_wait_asynccnt 0x0" ::: "memory");
# endif
#endif
}

__device__ __forceinline__ unsigned short f32_to_bf16_rne(float f) {
    unsigned int u = __float_as_uint(f);
    u += 0x7FFFu + ((u >> 16) & 1u);      // round-to-nearest-even
    return (unsigned short)(u >> 16);
}

// ---------------------------------------------------------------------------
// f32 -> bf16 bulk convert
// ---------------------------------------------------------------------------
__global__ void cvt_f32_bf16_kernel(const float* __restrict__ in,
                                    unsigned short* __restrict__ out, int n) {
    int i = blockIdx.x * blockDim.x + threadIdx.x;
    if (i < n) out[i] = f32_to_bf16_rne(in[i]);
}

// ---------------------------------------------------------------------------
// WMMA GEMM:  C[M,N] (f32) = A[M,K] (bf16 row-major) * W[N,K]^T (bf16 row-major)
// Block = 256 threads (8 waves), block tile 64(M) x 64(N), BK=32.
// Waves arranged 4(M) x 2(N); each wave owns a 16x32 strip: one A fragment
// feeds two v_wmma_f32_16x16x32_bf16. Tiles double-buffered in LDS with
// async global->LDS copies (each thread moves one b128 of A and of B per
// k-step; s_wait_asynccnt 2 synchronizes the in-flight prefetch).
// LDS rows padded to 40 bf16 (80B) to break ds_load_b128 bank conflicts.
// ---------------------------------------------------------------------------
__global__ void wmma_gemm_bf16_kernel(const unsigned short* __restrict__ A,
                                      const unsigned short* __restrict__ Bw,
                                      float* __restrict__ C,
                                      int M, int N, int K) {
    constexpr int BK = 32;
    constexpr int LR = 40;   // padded LDS row length (elements)
    __shared__ alignas(16) unsigned short ldsA[2][64 * LR];
    __shared__ alignas(16) unsigned short ldsB[2][64 * LR];

    const int tid  = threadIdx.x;
    const int lane = tid & 31;
    const int wave = tid >> 5;
    const int m0 = blockIdx.y * 64;
    const int n0 = blockIdx.x * 64;

    // ---- cooperative copy mapping: thread -> (row 0..63, col8 0/8/16/24) ----
    const int cr = tid >> 2;
    const int cc = (tid & 3) * 8;
    const unsigned short* gA = A  + (size_t)(m0 + cr) * K + cc;
    const unsigned short* gB = Bw + (size_t)(n0 + cr) * K + cc;
    unsigned short* lA0 = &ldsA[0][cr * LR + cc];
    unsigned short* lA1 = &ldsA[1][cr * LR + cc];
    unsigned short* lB0 = &ldsB[0][cr * LR + cc];
    unsigned short* lB1 = &ldsB[1][cr * LR + cc];

    auto issue_copy = [&](int buf, int k0) {
        unsigned short* la = buf ? lA1 : lA0;
        unsigned short* lb = buf ? lB1 : lB0;
#if USE_ASYNC_LDS
        __builtin_amdgcn_global_load_async_to_lds_b128(
            (AS1 v4i_vs*)(gA + k0), (AS3 v4i_vs*)la, 0, 0);
        __builtin_amdgcn_global_load_async_to_lds_b128(
            (AS1 v4i_vs*)(gB + k0), (AS3 v4i_vs*)lb, 0, 0);
#else
        *(uint4*)la = *(const uint4*)(gA + k0);
        *(uint4*)lb = *(const uint4*)(gB + k0);
#endif
    };

    // ---- per-wave compute mapping ----
    const int half = lane >> 4;     // 0: lanes 0-15, 1: lanes 16-31
    const int l16  = lane & 15;
    const int tm = (wave & 3) * 16; // wave grid 4(M) x 2(N)
    const int tn = (wave >> 2) * 32;

    v8f acc0 = {}, acc1 = {};
    const int S = K / BK;

    issue_copy(0, 0);
    for (int s = 0; s < S; ++s) {
        const int cur = s & 1;
        if (s + 1 < S) { issue_copy(cur ^ 1, (s + 1) * BK); async_wait_le2(); }
        else           { async_wait_0(); }
        __syncthreads();   // buffer `cur` fully resident for all waves

        FragU fa, fb0, fb1;
        // A 16x32 bf16: lanes 0-15 hold K 0-7/16-23, lanes 16-31 K 8-15/24-31
        const unsigned short* ar = (cur ? ldsA[1] : ldsA[0]) + (tm + l16) * LR;
        fa.q[0] = *(const uint4*)(ar + half * 8);
        fa.q[1] = *(const uint4*)(ar + 16 + half * 8);
        // B 32x16 bf16: lanes 0-15 hold K 0-7/16-23 of col n, lanes 16-31 K 8-15/24-31
        const unsigned short* bb = (cur ? ldsB[1] : ldsB[0]) + half * 8;
        const unsigned short* br0 = bb + (tn + l16) * LR;
        const unsigned short* br1 = bb + (tn + 16 + l16) * LR;
        fb0.q[0] = *(const uint4*)br0;  fb0.q[1] = *(const uint4*)(br0 + 16);
        fb1.q[0] = *(const uint4*)br1;  fb1.q[1] = *(const uint4*)(br1 + 16);

        acc0 = __builtin_amdgcn_wmma_f32_16x16x32_bf16(
            false, fa.v, false, fb0.v, (short)0, acc0, false, false);
        acc1 = __builtin_amdgcn_wmma_f32_16x16x32_bf16(
            false, fa.v, false, fb1.v, (short)0, acc1, false, false);

        __syncthreads();   // all reads done before buffer `cur` is rewritten
    }

    // D 16x16 f32: VGPR r -> row (r + 8*half), col l16
    float* c0 = C + (size_t)(m0 + tm + half * 8) * N + (n0 + tn + l16);
#pragma unroll
    for (int r = 0; r < 8; ++r) {
        c0[(size_t)r * N]      = acc0[r];
        c0[(size_t)r * N + 16] = acc1[r];
    }
}

// ---------------------------------------------------------------------------
// Depthwise causal conv (K=4) + bias + SiLU.
// x_in is the first D_INNER columns of xz (row stride N_XZ).
// ---------------------------------------------------------------------------
__global__ void conv_silu_kernel(const float* __restrict__ xz,
                                 const float* __restrict__ conv_w,
                                 const float* __restrict__ conv_b,
                                 float* __restrict__ x_conv) {
    int idx = blockIdx.x * blockDim.x + threadIdx.x;   // ROWS * D_INNER threads
    int d   = idx & (D_INNER - 1);
    int row = idx >> 11;                               // D_INNER = 2048
    int l   = row & (SEQ - 1);

    const float* w = conv_w + (size_t)d * 4;
    float s = conv_b[d];
#pragma unroll
    for (int j = 0; j < 4; ++j) {
        int lj = l + j - 3;
        if (lj >= 0)
            s += w[j] * xz[(size_t)(row + j - 3) * N_XZ + d];
    }
    x_conv[idx] = s / (1.0f + expf(-s));               // SiLU
}

// ---------------------------------------------------------------------------
// x_proj: ssm_in[row, n] = dot(x_conv[row,:], W_xproj[n,:]), n in [0,33).
// One wave per (n, row); shuffle reduction (wave32).
// ---------------------------------------------------------------------------
__global__ void xproj_kernel(const float* __restrict__ x_conv,
                             const float* __restrict__ W_xproj,
                             float* __restrict__ ssm_in) {
    const int n    = blockIdx.x;   // 0..32
    const int row  = blockIdx.y;   // 0..4095
    const int lane = threadIdx.x;  // 32 threads

    const float* xr = x_conv  + (size_t)row * D_INNER;
    const float* wr = W_xproj + (size_t)n   * D_INNER;
    float s = 0.0f;
    for (int k = lane; k < D_INNER; k += 32)
        s += xr[k] * wr[k];
#pragma unroll
    for (int o = 16; o > 0; o >>= 1)
        s += __shfl_down(s, o, 32);
    if (lane == 0)
        ssm_in[(size_t)row * N_PROJ + n] = s;
}

// ---------------------------------------------------------------------------
// Sequential selective scan. One lane owns one (b, d) channel; 16-state
// recurrence lives in registers; per-timestep ssm_in row staged through LDS.
// Fuses: delta = softplus(dt_raw*W_dt[d] + b_dt[d]),  h = exp(delta*A)h + dBu,
// y = <h,C> + x_conv*D,  y *= SiLU(z).  Emits y as bf16 for the out-proj GEMM.
// ---------------------------------------------------------------------------
__global__ void scan_kernel(const float* __restrict__ xz,
                            const float* __restrict__ x_conv,
                            const float* __restrict__ ssm_in,
                            const float* __restrict__ W_dt,
                            const float* __restrict__ b_dt,
                            const float* __restrict__ A_log,
                            const float* __restrict__ Dp,
                            unsigned short* __restrict__ y_bf16) {
    const int b = blockIdx.y;
    const int d = blockIdx.x * blockDim.x + threadIdx.x;   // 0..D_INNER-1

    __shared__ float srow[N_PROJ];

    float A[D_STATE], h[D_STATE];
#pragma unroll
    for (int s = 0; s < D_STATE; ++s) {
        A[s] = -expf(A_log[(size_t)d * D_STATE + s]);
        h[s] = 0.0f;
    }
    const float wdt = W_dt[d];
    const float bdt = b_dt[d];
    const float Dd  = Dp[d];

    for (int l = 0; l < SEQ; ++l) {
        const int row = b * SEQ + l;
        if (threadIdx.x < N_PROJ)
            srow[threadIdx.x] = ssm_in[(size_t)row * N_PROJ + threadIdx.x];
        __syncthreads();

        const float xc  = x_conv[(size_t)row * D_INNER + d];
        const float t   = srow[0] * wdt + bdt;
        const float dlt = (t > 20.0f) ? t : log1pf(expf(t));

        float y = 0.0f;
#pragma unroll
        for (int s = 0; s < D_STATE; ++s) {
            const float dA = expf(dlt * A[s]);
            h[s] = dA * h[s] + (dlt * srow[1 + s]) * xc;
            y += h[s] * srow[1 + D_STATE + s];
        }
        y += xc * Dd;

        const float z = xz[(size_t)row * N_XZ + D_INNER + d];
        y *= z / (1.0f + expf(-z));                        // * SiLU(z)

        y_bf16[(size_t)row * D_INNER + d] = f32_to_bf16_rne(y);
        __syncthreads();   // protect srow before next iteration's reload
    }
}

// ---------------------------------------------------------------------------
// Launcher
// ---------------------------------------------------------------------------
extern "C" void kernel_launch(void* const* d_in, const int* in_sizes, int n_in,
                              void* d_out, int out_size, void* d_ws, size_t ws_size,
                              hipStream_t stream) {
    const float* x       = (const float*)d_in[0];   // (B,L,D_MODEL)
    const float* W_in    = (const float*)d_in[1];   // (2*D_INNER, D_MODEL)
    const float* conv_w  = (const float*)d_in[2];   // (D_INNER,1,4)
    const float* conv_b  = (const float*)d_in[3];   // (D_INNER)
    const float* W_xproj = (const float*)d_in[4];   // (33, D_INNER)
    const float* W_dt    = (const float*)d_in[5];   // (D_INNER,1)
    const float* b_dt    = (const float*)d_in[6];   // (D_INNER)
    const float* A_log   = (const float*)d_in[7];   // (D_INNER,16)
    const float* Dp      = (const float*)d_in[8];   // (D_INNER)
    const float* W_out   = (const float*)d_in[9];   // (D_MODEL, D_INNER)
    float* out = (float*)d_out;                     // (B,L,D_MODEL)

    // ---- workspace carve-up (all offsets 256B-aligned) ----
    char* ws = (char*)d_ws;
    size_t cur = 0;
    auto carve = [&](size_t bytes) -> char* {
        char* p = ws + cur;
        cur += (bytes + 255) & ~(size_t)255;
        return p;
    };
    float*          xz        = (float*)carve((size_t)ROWS * N_XZ * 4);          // 64 MB
    float*          x_conv    = (float*)carve((size_t)ROWS * D_INNER * 4);       // 32 MB
    float*          ssm_in    = (float*)carve((size_t)ROWS * N_PROJ * 4);        // 0.5 MB
    unsigned short* y_bf16    = (unsigned short*)carve((size_t)ROWS * D_INNER * 2);   // 16 MB
    unsigned short* x_bf16    = (unsigned short*)carve((size_t)ROWS * D_MODEL * 2);   // 8 MB
    unsigned short* Win_bf16  = (unsigned short*)carve((size_t)N_XZ * D_MODEL * 2);   // 8 MB
    unsigned short* Wout_bf16 = (unsigned short*)carve((size_t)D_MODEL * D_INNER * 2);// 4 MB
    (void)ws_size; (void)n_in; (void)in_sizes; (void)out_size;

    // 1) bf16 conversions for the two large GEMMs
    {
        int n = ROWS * D_MODEL;
        cvt_f32_bf16_kernel<<<(n + 255) / 256, 256, 0, stream>>>(x, x_bf16, n);
        n = N_XZ * D_MODEL;
        cvt_f32_bf16_kernel<<<(n + 255) / 256, 256, 0, stream>>>(W_in, Win_bf16, n);
        n = D_MODEL * D_INNER;
        cvt_f32_bf16_kernel<<<(n + 255) / 256, 256, 0, stream>>>(W_out, Wout_bf16, n);
    }

    // 2) in-projection: xz[4096,4096] = x[4096,1024] @ W_in^T   (WMMA bf16)
    {
        dim3 grid(N_XZ / 64, ROWS / 64);
        wmma_gemm_bf16_kernel<<<grid, 256, 0, stream>>>(x_bf16, Win_bf16, xz,
                                                        ROWS, N_XZ, D_MODEL);
    }

    // 3) depthwise causal conv + bias + SiLU
    {
        int n = ROWS * D_INNER;
        conv_silu_kernel<<<n / 256, 256, 0, stream>>>(xz, conv_w, conv_b, x_conv);
    }

    // 4) x_proj (33 outputs/row, shuffle-reduced dot products)
    {
        dim3 grid(N_PROJ, ROWS);
        xproj_kernel<<<grid, 32, 0, stream>>>(x_conv, W_xproj, ssm_in);
    }

    // 5) recurrent selective scan, fused gating, bf16 output
    {
        dim3 grid(D_INNER / 256, BATCH);
        scan_kernel<<<grid, 256, 0, stream>>>(xz, x_conv, ssm_in,
                                              W_dt, b_dt, A_log, Dp, y_bf16);
    }

    // 6) out-projection: out[4096,1024] = y[4096,2048] @ W_out^T   (WMMA bf16)
    {
        dim3 grid(D_MODEL / 64, ROWS / 64);
        wmma_gemm_bf16_kernel<<<grid, 256, 0, stream>>>(y_bf16, Wout_bf16, out,
                                                        ROWS, D_MODEL, D_INNER);
    }
}
